// MultiHeadedDepthAttention_10926396801349
// MI455X (gfx1250) — matrix operand order repaired and checked by reading, not compiled, not run on hardware
//
#include <hip/hip_runtime.h>
#include <hip/hip_bf16.h>
#include <stdint.h>

typedef __attribute__((ext_vector_type(16))) _Float16 v16h;
typedef __attribute__((ext_vector_type(8)))  float    v8f;
typedef __attribute__((ext_vector_type(4)))  uint32_t v4u;
typedef __attribute__((ext_vector_type(8)))  int      v8i_t;
typedef __attribute__((ext_vector_type(4)))  int      v4i_t;

// ---------------- WMMA helpers (CDNA5 16x16x32 f16 -> f32) ----------------

__device__ __forceinline__ v8f wmma_f16(v16h a, v16h b, v8f c) {
  return __builtin_amdgcn_wmma_f32_16x16x32_f16(
      /*neg_a=*/false, a, /*neg_b=*/false, b,
      /*c_mod=*/(short)0, c, /*reuse_a=*/false, /*reuse_b=*/false);
}

// A-matrix 16x32 (MxK), row-major source with leading dim ld (elements).
__device__ __forceinline__ v16h load_a_frag(const _Float16* A, int ld, int lane) {
  int row = lane & 15;
  int kb  = (lane >> 4) << 4;
  const _Float16* p = A + (size_t)row * ld;
  v16h r;
#pragma unroll
  for (int v = 0; v < 8; ++v) {
    int base = kb + (v << 1);
    r[2 * v]     = p[base];
    r[2 * v + 1] = p[base + 1];
  }
  return r;
}

// B-matrix 32x16 (KxN), row-major source B[k][n] with leading dim ld.
__device__ __forceinline__ v16h load_b_frag(const _Float16* B, int ld, int lane) {
  int n  = lane & 15;
  int k0 = (lane >> 4) << 4;
  v16h r;
#pragma unroll
  for (int v = 0; v < 16; ++v) r[v] = B[(size_t)(k0 + v) * ld + n];
  return r;
}

// B = K^T where source K is row-major (N x Kdim): B[k][n] = Ksrc[n][k].
__device__ __forceinline__ v16h load_bt_frag(const _Float16* Ksrc, int ld, int lane) {
  int n  = lane & 15;
  int k0 = (lane >> 4) << 4;
  const _Float16* p = Ksrc + (size_t)n * ld + k0;
  v16h r;
#pragma unroll
  for (int v = 0; v < 16; ++v) r[v] = p[v];
  return r;
}

// ---------------- CDNA5 async global->LDS copy (ASYNCcnt path) ----------------
__device__ __forceinline__ void async_load_b128(uint32_t lds_off, const void* gaddr) {
  asm volatile("global_load_async_to_lds_b128 %0, %1, off"
               :: "v"(lds_off), "v"(gaddr) : "memory");
}
__device__ __forceinline__ void wait_async0() {
  asm volatile("s_wait_asynccnt 0" ::: "memory");
}
__device__ __forceinline__ void wait_async4() {  // leave the 4 just-issued pending
  asm volatile("s_wait_asynccnt 4" ::: "memory");
}

// ---------------- CDNA5 Tensor Data Mover: 2D f16 tile -> LDS ----------------
// tile_d0 contiguous halfs per row, tile_d1 rows, row stride in elements.
// D# layout per CDNA5 ISA.
__device__ __forceinline__ void tdm_load_2d_f16(uint32_t lds_off, const void* gaddr,
                                                uint32_t tile_d0, uint32_t tile_d1,
                                                uint64_t row_stride_elems) {
  uint64_t ga = (uint64_t)(uintptr_t)gaddr;
  v4u g0;
  g0[0] = 1u;                                        // count=1, user mode, no gather
  g0[1] = lds_off;                                   // lds_addr (bytes)
  g0[2] = (uint32_t)(ga & 0xffffffffu);              // global_addr[31:0]
  g0[3] = (uint32_t)((ga >> 32) & 0x01ffffffu) | (2u << 30);  // addr[56:32] | type=2
  v8i_t g1;
  g1[0] = (int)(1u << 16);                           // workgroup_mask=0, data_size=1 (2B)
  g1[1] = (int)((tile_d1 & 0u) | ((tile_d0 & 0xffffu) << 16));        // tensor_dim0 lo16
  g1[2] = (int)(((tile_d0 >> 16) & 0xffffu) | ((tile_d1 & 0xffffu) << 16)); // td0 hi / td1 lo
  g1[3] = (int)(((tile_d1 >> 16) & 0xffffu) | ((tile_d0 & 0xffffu) << 16)); // td1 hi / tile_dim0
  g1[4] = (int)(tile_d1 & 0xffffu);                  // tile_dim1, tile_dim2=0
  g1[5] = (int)(uint32_t)(row_stride_elems & 0xffffffffu);   // tensor_dim0_stride[31:0]
  g1[6] = (int)(uint32_t)((row_stride_elems >> 32) & 0xffffu); // stride[47:32], dim1_stride lo=0
  g1[7] = 0;
  v4i_t z4 = {0, 0, 0, 0};
#if __clang_major__ >= 23
  v8i_t z8 = {0, 0, 0, 0, 0, 0, 0, 0};
  __builtin_amdgcn_tensor_load_to_lds(g0, g1, z4, z4, z8, 0);
#else
  __builtin_amdgcn_tensor_load_to_lds(g0, g1, z4, z4, 0);
#endif
}

// ---------------- Kernel 1: QKV 1x1-conv as WMMA GEMM ----------------
__global__ void qkv_proj_kernel(const float* __restrict__ x,
                                const float* __restrict__ wq, const float* __restrict__ bq,
                                const float* __restrict__ wk, const float* __restrict__ bk,
                                const float* __restrict__ wv, const float* __restrict__ bv,
                                _Float16* __restrict__ q16, _Float16* __restrict__ k16,
                                _Float16* __restrict__ v16o) {
  __shared__ _Float16 Wl[128 * 128];
  int which = blockIdx.y;
  const float* Wsrc = (which == 0) ? wq : (which == 1) ? wk : wv;
  const float* bias = (which == 0) ? bq : (which == 1) ? bk : bv;
  _Float16* dst = (which == 0) ? q16 : (which == 1) ? k16 : v16o;

  for (int i = threadIdx.x; i < 128 * 128; i += blockDim.x)
    Wl[i] = (_Float16)Wsrc[i];
  __syncthreads();

  int wave = threadIdx.x >> 5, lane = threadIdx.x & 31;
  int mBase = wave * 16;
  long colBase = (long)blockIdx.x * 16;

  int nlane = lane & 15;
  long col = colBase + nlane;
  int nImg = (int)(col >> 12);
  int hw   = (int)(col & 4095);
  const float* xb = x + ((size_t)nImg * 128) * 4096 + hw;
  int kk0 = (lane >> 4) << 4;

  v8f acc = {};
  for (int k0 = 0; k0 < 128; k0 += 32) {
    v16h a = load_a_frag(Wl + (size_t)mBase * 128 + k0, 128, lane);
    v16h b;
#pragma unroll
    for (int v = 0; v < 16; ++v) b[v] = (_Float16)xb[(size_t)(k0 + kk0 + v) * 4096];
    acc = wmma_f16(a, b, acc);
  }
  int mo = (lane >> 4) * 8;
  for (int rr = 0; rr < 8; ++rr) {
    int o = mBase + mo + rr;
    dst[(size_t)o * 65536 + col] = (_Float16)(acc[rr] + bias[o]);
  }
}

// ---------------- Kernel 2: depthwise 3x3 add into V ----------------
__global__ void vle_add_kernel(const float* __restrict__ x, const float* __restrict__ wvle,
                               const float* __restrict__ bvle, _Float16* __restrict__ v16o) {
  size_t idx = (size_t)blockIdx.x * blockDim.x + threadIdx.x;
  if (idx >= (size_t)128 * 65536) return;
  int c = (int)(idx >> 16), col = (int)(idx & 65535);
  int n = col >> 12, hw = col & 4095, h = hw >> 6, w = hw & 63;
  const float* xb = x + ((size_t)n * 128 + c) * 4096;
  const float* wf = wvle + c * 9;
  float s = bvle[c];
#pragma unroll
  for (int kh = 0; kh < 3; ++kh)
#pragma unroll
    for (int kw = 0; kw < 3; ++kw) {
      int h2 = h + kh - 1, w2 = w + kw - 1;
      if ((unsigned)h2 < 64u && (unsigned)w2 < 64u) s += wf[kh * 3 + kw] * xb[h2 * 64 + w2];
    }
  v16o[idx] = (_Float16)((float)v16o[idx] + s);
}

// ---------------- Kernel 3: fused 2-layer depth CNN (256^2 -> 64^2) ----------------
__global__ void depth_kernel(const float* __restrict__ dm,
                             const float* __restrict__ wd1, const float* __restrict__ bd1,
                             const float* __restrict__ wd2, const float* __restrict__ bd2,
                             float* __restrict__ dout) {
  int idx = blockIdx.x * blockDim.x + threadIdx.x;
  if (idx >= 16 * 4096) return;
  int n = idx >> 12, hw = idx & 4095, y = hw >> 6, xk = hw & 63;
  const float* dmb = dm + (size_t)n * 65536;
  float acc = bd2[0];
  for (int c = 0; c < 128; ++c) {
    const float* w1 = wd1 + c * 9;
    float b1 = bd1[c];
#pragma unroll
    for (int kh = 0; kh < 3; ++kh)
#pragma unroll
      for (int kw = 0; kw < 3; ++kw) {
        int u = 2 * y + kh - 1, v = 2 * xk + kw - 1;
        if ((unsigned)u < 128u && (unsigned)v < 128u) {
          float d1 = b1;
#pragma unroll
          for (int ih = 0; ih < 3; ++ih)
#pragma unroll
            for (int iw = 0; iw < 3; ++iw) {
              int p = 2 * u + ih - 1, q = 2 * v + iw - 1;
              if ((unsigned)p < 256u && (unsigned)q < 256u) d1 += w1[ih * 3 + iw] * dmb[p * 256 + q];
            }
          d1 = d1 > 0.f ? d1 : 0.f;
          acc += wd2[c * 9 + kh * 3 + kw] * d1;
        }
      }
  }
  dout[idx] = acc > 0.f ? acc : 0.f;
}

// ---------------- Kernel 4: per-patch mask + depth stats ----------------
__global__ void patch_stats_kernel(const float* __restrict__ m, const float* __restrict__ dep,
                                   float* __restrict__ maskf, float* __restrict__ sig,
                                   int pw, int ph, int out_w, int out_h, int L) {
  int idx = blockIdx.x * blockDim.x + threadIdx.x;
  if (idx >= 2 * L) return;
  int bb = idx / L, l = idx % L;
  int per = out_h * out_w;
  int tt = l / per, r = l % per, oh = r / out_w, ow = r % out_w;
  int n = bb * 8 + tt;
  int h0 = oh * ph, w0 = ow * pw, hw = ph * pw;
  const float* mb = m + (size_t)n * 4096;
  const float* db = dep + (size_t)n * 4096;
  float vals[64];
  float msum = 0.f;
  for (int i = 0; i < hw; ++i) {
    int hh = h0 + i / pw, ww = w0 + i % pw;
    msum += mb[hh * 64 + ww];
    vals[i] = db[hh * 64 + ww];
  }
  maskf[idx] = (msum / hw > 0.5f) ? 1.f : 0.f;
  for (int i = 1; i < hw; ++i) {
    float v = vals[i];
    int j = i - 1;
    while (j >= 0 && vals[j] > v) { vals[j + 1] = vals[j]; --j; }
    vals[j + 1] = v;
  }
  float dmin = vals[0], dmed = vals[(hw - 1) / 2], dmax = vals[hw - 1];
  sig[(size_t)0 * 2 * L + idx] = 1.f / (1.f + __expf(-dmed));
  sig[(size_t)1 * 2 * L + idx] = 1.f / (1.f + __expf(-dmax));
  sig[(size_t)2 * 2 * L + idx] = 1.f / (1.f + __expf(-dmin));
}

// ---------------- Kernel 5: gather patched Q/K/V (row-major L x D, f16) ----------------
__global__ void patch_qkv_kernel(const _Float16* __restrict__ q16, const _Float16* __restrict__ k16,
                                 const _Float16* __restrict__ v16i,
                                 _Float16* __restrict__ Qp, _Float16* __restrict__ Kp,
                                 _Float16* __restrict__ Vp,
                                 int scaleIdx, int pw, int ph, int out_w, int out_h, int L, int Dd) {
  size_t idx = (size_t)blockIdx.x * blockDim.x + threadIdx.x;
  size_t total = (size_t)2 * L * Dd;
  if (idx >= total) return;
  int e = (int)(idx % Dd);
  size_t r = idx / Dd;
  int l = (int)(r % L);
  int bb = (int)(r / L);
  int per = out_h * out_w;
  int tt = l / per, r2 = l % per, oh = r2 / out_w, ow = r2 % out_w;
  int phw = ph * pw;
  int ch = e / phw, r3 = e % phw, pph = r3 / pw, ppw = r3 % pw;
  int c = scaleIdx * 64 + ch;
  int n = bb * 8 + tt;
  int hh = oh * ph + pph, ww = ow * pw + ppw;
  size_t src = (size_t)c * 65536 + (size_t)n * 4096 + hh * 64 + ww;
  Qp[idx] = q16[src];
  Kp[idx] = k16[src];
  Vp[idx] = v16i[src];
}

// ---------------- Kernel 6: S = scale * Q K^T, TDM-staged LDS WMMA ----------------
// Block tile 128x128, K-panel 32. Panels DMA'd by the Tensor Data Mover (wave 0),
// tracked with TENSORcnt.
__global__ void s_gemm_tiled(const _Float16* __restrict__ Qp, const _Float16* __restrict__ Kp,
                             float* __restrict__ S, int L, int D, float scale) {
  __shared__ _Float16 Alds[128 * 32];   // Q rows  (m x k)
  __shared__ _Float16 Blds[128 * 32];   // K rows  (n x k)
  int bb = blockIdx.z;
  int mBase = blockIdx.x * 128, nBase = blockIdx.y * 128;
  const _Float16* Qb = Qp + (size_t)bb * L * D + (size_t)mBase * D;
  const _Float16* Kb = Kp + (size_t)bb * L * D + (size_t)nBase * D;
  int tid = threadIdx.x, lane = tid & 31, wave = tid >> 5;
  int mW = (wave >> 2) * 64, nW = (wave & 3) * 32;
  v8f acc[4][2] = {};

  for (int k0 = 0; k0 < D; k0 += 32) {
    __syncthreads();                    // previous compute done before overwrite
    if (tid < 32) {                     // one wave drives the TDM
      tdm_load_2d_f16((uint32_t)(uintptr_t)&Alds[0], Qb + k0, 32, 128, (uint64_t)D);
      tdm_load_2d_f16((uint32_t)(uintptr_t)&Blds[0], Kb + k0, 32, 128, (uint64_t)D);
      __builtin_amdgcn_s_wait_tensorcnt(0);
    }
    __syncthreads();
    v16h bf[2];
#pragma unroll
    for (int ni = 0; ni < 2; ++ni)
      bf[ni] = load_bt_frag(&Blds[(nW + ni * 16) * 32], 32, lane);
#pragma unroll
    for (int mi = 0; mi < 4; ++mi) {
      v16h af = load_a_frag(&Alds[(mW + mi * 16) * 32], 32, lane);
#pragma unroll
      for (int ni = 0; ni < 2; ++ni) acc[mi][ni] = wmma_f16(af, bf[ni], acc[mi][ni]);
    }
  }

  float* Sb = S + (size_t)bb * L * L;
  int n = lane & 15, mo = (lane >> 4) * 8;
  for (int mi = 0; mi < 4; ++mi)
    for (int ni = 0; ni < 2; ++ni)
      for (int rr = 0; rr < 8; ++rr)
        Sb[(size_t)(mBase + mW + mi * 16 + mo + rr) * L + nBase + nW + ni * 16 + n] =
            acc[mi][ni][rr] * scale;
}

// ---------------- Kernel 7: masked/biased row softmax -> f16 P ----------------
__global__ void softmax_kernel(const float* __restrict__ S, const float* __restrict__ maskf,
                               const float* __restrict__ sig, _Float16* __restrict__ P,
                               int L, int headSel) {
  int row = blockIdx.x;
  int bb = row / L;
  const float* Srow = S + (size_t)row * L;
  const float* mrow = maskf + (size_t)bb * L;
  const float* grow = sig + ((size_t)headSel * 2 + bb) * L;
  __shared__ float redMax[8];
  __shared__ float redSum[8];
  int tid = threadIdx.x, lane = tid & 31, wv = tid >> 5;

  float mx = -3.0e38f;
  for (int k = tid; k < L; k += 256) {
    float z = (mrow[k] > 0.5f) ? -1.0e9f : Srow[k] * grow[k];
    mx = fmaxf(mx, z);
  }
#pragma unroll
  for (int o = 16; o > 0; o >>= 1) mx = fmaxf(mx, __shfl_xor(mx, o, 32));
  if (lane == 0) redMax[wv] = mx;
  __syncthreads();
  mx = fmaxf(fmaxf(fmaxf(redMax[0], redMax[1]), fmaxf(redMax[2], redMax[3])),
             fmaxf(fmaxf(redMax[4], redMax[5]), fmaxf(redMax[6], redMax[7])));

  float sm = 0.f;
  for (int k = tid; k < L; k += 256) {
    float z = (mrow[k] > 0.5f) ? -1.0e9f : Srow[k] * grow[k];
    sm += __expf(z - mx);
  }
#pragma unroll
  for (int o = 16; o > 0; o >>= 1) sm += __shfl_xor(sm, o, 32);
  if (lane == 0) redSum[wv] = sm;
  __syncthreads();
  sm = redSum[0] + redSum[1] + redSum[2] + redSum[3] +
       redSum[4] + redSum[5] + redSum[6] + redSum[7];
  float inv = 1.0f / sm;

  _Float16* Prow = P + (size_t)row * L;
  for (int k = tid; k < L; k += 256) {
    float z = (mrow[k] > 0.5f) ? -1.0e9f : Srow[k] * grow[k];
    Prow[k] = (_Float16)(__expf(z - mx) * inv);
  }
}

// ---------------- Kernel 8: Y += alpha * P V, double-buffered async WMMA ----------------
__device__ __forceinline__ void pv_stage(const _Float16* Pb, const _Float16* Vb,
                                         _Float16* Abuf, _Float16* Bbuf,
                                         int L, int D, int k0, int tid) {
#pragma unroll
  for (int it = 0; it < 2; ++it) {
    int chunk = tid + it * 256;
    int rowA = chunk >> 2, c4 = chunk & 3;
    async_load_b128((uint32_t)(uintptr_t)&Abuf[rowA * 32 + c4 * 8],
                    Pb + (size_t)rowA * L + k0 + c4 * 8);
    int rowB = chunk >> 4, c16 = chunk & 15;
    async_load_b128((uint32_t)(uintptr_t)&Bbuf[rowB * 128 + c16 * 8],
                    Vb + (size_t)(k0 + rowB) * D + c16 * 8);
  }
}

__global__ void pv_gemm_tiled(const _Float16* __restrict__ P, const _Float16* __restrict__ Vp,
                              float* __restrict__ Y, int L, int D, float alpha, int accumulate) {
  __shared__ _Float16 Alds[2][128 * 32];   // P rows (m x k), double buffered
  __shared__ _Float16 Blds[2][32 * 128];   // V rows (k x n), double buffered
  int bb = blockIdx.z;
  int mBase = blockIdx.x * 128, nBase = blockIdx.y * 128;
  const _Float16* Pb = P + (size_t)bb * L * L + (size_t)mBase * L;
  const _Float16* Vb = Vp + (size_t)bb * L * D + nBase;
  int tid = threadIdx.x, lane = tid & 31, wave = tid >> 5;
  int mW = (wave >> 2) * 64, nW = (wave & 3) * 32;
  v8f acc[4][2] = {};
  int steps = L >> 5;

  pv_stage(Pb, Vb, Alds[0], Blds[0], L, D, 0, tid);   // prologue
  for (int i = 0; i < steps; ++i) {
    int cur = i & 1;
    if (i + 1 < steps) {
      pv_stage(Pb, Vb, Alds[cur ^ 1], Blds[cur ^ 1], L, D, (i + 1) << 5, tid);
      wait_async4();                  // stage i complete; stage i+1 in flight
    } else {
      wait_async0();
    }
    __syncthreads();
    v16h bf[2];
#pragma unroll
    for (int ni = 0; ni < 2; ++ni)
      bf[ni] = load_b_frag(&Blds[cur][nW + ni * 16], 128, lane);
#pragma unroll
    for (int mi = 0; mi < 4; ++mi) {
      v16h af = load_a_frag(&Alds[cur][(mW + mi * 16) * 32], 32, lane);
#pragma unroll
      for (int ni = 0; ni < 2; ++ni) acc[mi][ni] = wmma_f16(af, bf[ni], acc[mi][ni]);
    }
    __syncthreads();                  // all reads done before buffer reuse
  }

  float* Yb = Y + (size_t)bb * L * D;
  int n = lane & 15, mo = (lane >> 4) * 8;
  for (int mi = 0; mi < 4; ++mi)
    for (int ni = 0; ni < 2; ++ni)
      for (int rr = 0; rr < 8; ++rr) {
        size_t idx = (size_t)(mBase + mW + mi * 16 + mo + rr) * D + nBase + nW + ni * 16 + n;
        float prev = accumulate ? Yb[idx] : 0.f;
        Yb[idx] = prev + alpha * acc[mi][ni][rr];
      }
}

// ---------------- Kernel 9: scatter Y back to feature map (f16 [c][n][hw]) -----------
__global__ void unpatch_kernel(const float* __restrict__ Y, _Float16* __restrict__ F,
                               int scaleIdx, int pw, int ph, int out_w, int out_h, int L, int Dd) {
  size_t idx = (size_t)blockIdx.x * blockDim.x + threadIdx.x;
  size_t total = (size_t)2 * L * Dd;
  if (idx >= total) return;
  int e = (int)(idx % Dd);
  size_t r = idx / Dd;
  int l = (int)(r % L);
  int bb = (int)(r / L);
  int per = out_h * out_w;
  int tt = l / per, r2 = l % per, oh = r2 / out_w, ow = r2 % out_w;
  int phw = ph * pw;
  int ch = e / phw, r3 = e % phw, pph = r3 / pw, ppw = r3 % pw;
  int c = scaleIdx * 64 + ch;
  int n = bb * 8 + tt;
  int hh = oh * ph + pph, ww = ow * pw + ppw;
  F[(size_t)c * 65536 + (size_t)n * 4096 + hh * 64 + ww] = (_Float16)Y[idx];
}

// ---------------- Kernel 10: 3x3 output conv as implicit WMMA GEMM + LeakyReLU --------
__global__ void out_conv_kernel(const _Float16* __restrict__ F, const float* __restrict__ wo,
                                const float* __restrict__ bo, float* __restrict__ out) {
  int wave = threadIdx.x >> 5, lane = threadIdx.x & 31;
  int mBase = wave * 16;
  int n = blockIdx.y;
  int pBase = blockIdx.x * 16;
  int nlane = lane & 15;
  int p = pBase + nlane;
  int hh = p >> 6, ww = p & 63;
  int rowA = mBase + (lane & 15);
  int kb = (lane >> 4) << 4;
  int khalf = (lane >> 4) << 4;

  v8f acc = {};
  for (int k0 = 0; k0 < 1152; k0 += 32) {
    __builtin_prefetch(wo + (size_t)rowA * 1152 + k0 + 32, 0, 0);
    v16h a;
    {
      const float* pa = wo + (size_t)rowA * 1152 + k0;
#pragma unroll
      for (int v = 0; v < 8; ++v) {
        int base = kb + (v << 1);
        a[2 * v]     = (_Float16)pa[base];
        a[2 * v + 1] = (_Float16)pa[base + 1];
      }
    }
    v16h b;
    {
      int kk0 = k0 + khalf;
#pragma unroll
      for (int v = 0; v < 16; ++v) {
        int kk = kk0 + v;
        int c = kk / 9, r2 = kk % 9, kh = r2 / 3, kw = r2 % 3;
        int h2 = hh + kh - 1, w2 = ww + kw - 1;
        _Float16 val = (_Float16)0.f;
        if ((unsigned)h2 < 64u && (unsigned)w2 < 64u)
          val = F[(size_t)c * 65536 + (size_t)n * 4096 + h2 * 64 + w2];
        b[v] = val;
      }
    }
    acc = wmma_f16(a, b, acc);
  }
  int mo = (lane >> 4) * 8;
  for (int rr = 0; rr < 8; ++rr) {
    int o = mBase + mo + rr;
    float v = acc[rr] + bo[o];
    v = v >= 0.f ? v : 0.2f * v;
    out[((size_t)n * 128 + o) * 4096 + p] = v;
  }
}

// ---------------- Host launch ----------------
extern "C" void kernel_launch(void* const* d_in, const int* in_sizes, int n_in,
                              void* d_out, int out_size, void* d_ws, size_t ws_size,
                              hipStream_t stream) {
  (void)in_sizes; (void)n_in; (void)out_size; (void)ws_size;
  const float* x    = (const float*)d_in[0];
  const float* m    = (const float*)d_in[1];
  const float* dm   = (const float*)d_in[2];
  const float* wq   = (const float*)d_in[3];
  const float* bq   = (const float*)d_in[4];
  const float* wk   = (const float*)d_in[5];
  const float* bk   = (const float*)d_in[6];
  const float* wv   = (const float*)d_in[7];
  const float* bv   = (const float*)d_in[8];
  const float* wvle = (const float*)d_in[9];
  const float* bvle = (const float*)d_in[10];
  const float* wd1  = (const float*)d_in[11];
  const float* bd1  = (const float*)d_in[12];
  const float* wd2  = (const float*)d_in[13];
  const float* bd2  = (const float*)d_in[14];
  const float* wo   = (const float*)d_in[15];
  const float* bo   = (const float*)d_in[16];

  uint8_t* base = (uint8_t*)d_ws;
  auto alloc = [&](size_t bytes) { uint8_t* p = base; base += (bytes + 255) & ~(size_t)255; return p; };

  _Float16* q16  = (_Float16*)alloc((size_t)128 * 65536 * 2);
  _Float16* k16  = (_Float16*)alloc((size_t)128 * 65536 * 2);
  _Float16* v16b = (_Float16*)alloc((size_t)128 * 65536 * 2);
  float* dep     = (float*)alloc((size_t)16 * 4096 * 4);
  float* mask0   = (float*)alloc((size_t)2 * 2048 * 4);
  float* sig0    = (float*)alloc((size_t)3 * 2 * 2048 * 4);
  float* mask1   = (float*)alloc((size_t)2 * 512 * 4);
  float* sig1    = (float*)alloc((size_t)3 * 2 * 512 * 4);
  _Float16* Qp0  = (_Float16*)alloc((size_t)2 * 2048 * 1024 * 2);
  _Float16* Kp0  = (_Float16*)alloc((size_t)2 * 2048 * 1024 * 2);
  _Float16* Vp0  = (_Float16*)alloc((size_t)2 * 2048 * 1024 * 2);
  _Float16* Qp1  = (_Float16*)alloc((size_t)2 * 512 * 4096 * 2);
  _Float16* Kp1  = (_Float16*)alloc((size_t)2 * 512 * 4096 * 2);
  _Float16* Vp1  = (_Float16*)alloc((size_t)2 * 512 * 4096 * 2);
  float* S0      = (float*)alloc((size_t)2 * 2048 * 2048 * 4);
  float* S1      = (float*)alloc((size_t)2 * 512 * 512 * 4);
  _Float16* P0   = (_Float16*)alloc((size_t)2 * 2048 * 2048 * 2);
  _Float16* P1   = (_Float16*)alloc((size_t)2 * 512 * 512 * 2);
  float* Y0      = (float*)alloc((size_t)2 * 2048 * 1024 * 4);
  float* Y1      = (float*)alloc((size_t)2 * 512 * 4096 * 4);
  _Float16* F    = (_Float16*)alloc((size_t)128 * 65536 * 2);

  dim3 blk(256);

  qkv_proj_kernel<<<dim3(4096, 3), blk, 0, stream>>>(x, wq, bq, wk, bk, wv, bv, q16, k16, v16b);
  vle_add_kernel<<<32768, blk, 0, stream>>>(x, wvle, bvle, v16b);
  depth_kernel<<<256, blk, 0, stream>>>(dm, wd1, bd1, wd2, bd2, dep);

  patch_stats_kernel<<<16, blk, 0, stream>>>(m, dep, mask0, sig0, 4, 4, 16, 16, 2048);
  patch_stats_kernel<<<4, blk, 0, stream>>>(m, dep, mask1, sig1, 8, 8, 8, 8, 512);

  patch_qkv_kernel<<<16384, blk, 0, stream>>>(q16, k16, v16b, Qp0, Kp0, Vp0, 0, 4, 4, 16, 16, 2048, 1024);
  patch_qkv_kernel<<<16384, blk, 0, stream>>>(q16, k16, v16b, Qp1, Kp1, Vp1, 1, 8, 8, 8, 8, 512, 4096);

  s_gemm_tiled<<<dim3(16, 16, 2), blk, 0, stream>>>(Qp0, Kp0, S0, 2048, 1024, 1.0f / 32.0f);
  s_gemm_tiled<<<dim3(4, 4, 2), blk, 0, stream>>>(Qp1, Kp1, S1, 512, 4096, 1.0f / 64.0f);

  for (int hsel = 0; hsel < 3; ++hsel) {
    softmax_kernel<<<4096, blk, 0, stream>>>(S0, mask0, sig0, P0, 2048, hsel);
    pv_gemm_tiled<<<dim3(16, 8, 2), blk, 0, stream>>>(P0, Vp0, Y0, 2048, 1024, 1.0f / 3.0f, hsel > 0);
    softmax_kernel<<<1024, blk, 0, stream>>>(S1, mask1, sig1, P1, 512, hsel);
    pv_gemm_tiled<<<dim3(4, 32, 2), blk, 0, stream>>>(P1, Vp1, Y1, 512, 4096, 1.0f / 3.0f, hsel > 0);
  }

  unpatch_kernel<<<16384, blk, 0, stream>>>(Y0, F, 0, 4, 4, 16, 16, 2048, 1024);
  unpatch_kernel<<<16384, blk, 0, stream>>>(Y1, F, 1, 8, 8, 8, 8, 512, 4096);

  out_conv_kernel<<<dim3(256, 16), blk, 0, stream>>>(F, wo, bo, (float*)d_out);

  hipMemcpyAsync((float*)d_out + (size_t)16 * 128 * 4096, dm,
                 (size_t)16 * 65536 * sizeof(float), hipMemcpyDeviceToDevice, stream);
}
